// TitansMemoryLayer_29910152249560
// MI455X (gfx1250) — matrix operand order repaired and checked by reading, not compiled, not run on hardware
//
#include <hip/hip_runtime.h>
#include <hip/hip_bf16.h>

typedef __attribute__((ext_vector_type(16))) __bf16 v16bf;
typedef __attribute__((ext_vector_type(8)))  float  v8f;

#define B_      4
#define S_      4096
#define D_      1024
#define M_      4096
#define NTOK    (B_ * S_)            /* 16384 */
#define DECAY   0.9f
#define LNDECAY -0.105360516f        /* ln(0.9) */
#define CHUNK   128
#define NCHUNK  (S_ / CHUNK)         /* 32 */

__device__ __forceinline__ unsigned short f2bf(float f) {
    unsigned int u = __float_as_uint(f);
    u += 0x7FFFu + ((u >> 16) & 1u);          // round-to-nearest-even
    return (unsigned short)(u >> 16);
}

// ---------------------------------------------------------------------------
// EMA pass 1: per-(b,chunk,d) local scan; writes local states (f32, to be
// fixed up) and the chunk-final local value into carry[].
// ---------------------------------------------------------------------------
__global__ void ema_pass1(const float* __restrict__ x,
                          float* __restrict__ state,
                          float* __restrict__ carry) {
    int tid = blockIdx.x * blockDim.x + threadIdx.x;   // B*NCHUNK*D threads
    int d = tid % D_;
    int r = tid / D_;
    int j = r % NCHUNK;
    int b = r / NCHUNK;
    size_t base = ((size_t)b * S_ + (size_t)j * CHUNK) * D_ + d;
    float h = 0.0f;
    for (int t = 0; t < CHUNK; ++t) {
        h = DECAY * h + (1.0f - DECAY) * x[base + (size_t)t * D_];
        state[base + (size_t)t * D_] = h;
    }
    carry[((size_t)b * NCHUNK + j) * D_ + d] = h;
}

// ---------------------------------------------------------------------------
// EMA pass 2: per-(b,d) serial combine of the 32 chunk carries.
// carry[b][j][d] is rewritten in place with the carry-IN of chunk j.
// ---------------------------------------------------------------------------
__global__ void ema_pass2(float* __restrict__ carry) {
    int tid = blockIdx.x * blockDim.x + threadIdx.x;   // B*D threads
    int d = tid % D_;
    int b = tid / D_;
    const float f = __expf((float)CHUNK * LNDECAY);    // 0.9^128
    float c = 0.0f;
    for (int j = 0; j < NCHUNK; ++j) {
        size_t idx = ((size_t)b * NCHUNK + j) * D_ + d;
        float L = carry[idx];
        carry[idx] = c;                                // carry-in for chunk j
        c = L + f * c;
    }
}

// ---------------------------------------------------------------------------
// EMA pass 3: h = local + 0.9^(t+1) * carry_in. Writes final f32 state
// (3rd output) and bf16 state (GEMM1 A matrix) in one coalesced sweep.
// ---------------------------------------------------------------------------
__global__ void ema_fixup(float* __restrict__ state,
                          const float* __restrict__ carry,
                          unsigned short* __restrict__ stateBf) {
    size_t tid = (size_t)blockIdx.x * blockDim.x + threadIdx.x;
    int d = (int)(tid % D_);
    int s = (int)((tid / D_) % S_);
    int b = (int)(tid / ((size_t)D_ * S_));
    int j = s / CHUNK;
    int t = s % CHUNK;
    float cin = carry[((size_t)b * NCHUNK + j) * D_ + d];
    float v = state[tid] + __expf((float)(t + 1) * LNDECAY) * cin;
    state[tid]   = v;
    stateBf[tid] = f2bf(v);
}

__global__ void f32_to_bf16_kernel(const float* __restrict__ in,
                                   unsigned short* __restrict__ out, int n) {
    int i = blockIdx.x * blockDim.x + threadIdx.x;
    if (i < n) out[i] = f2bf(in[i]);
}

__global__ void zero_loss(float* loss) { *loss = 0.0f; }

// ---------------------------------------------------------------------------
// Tiled bf16 WMMA GEMM:  out[i,j] = sum_k A[i,k] * Bw[j,k]
//   A  : [Itot x K] bf16 row-major      Bw : [Jtot x K] bf16 row-major
// Block: 256 threads (8 wave32), tile 128(I) x 128(J), K-step 32,
// double-buffered LDS (one barrier per K-step).
// Each wave owns a 32x64 patch = 2x4 v_wmma_f32_16x16x32_bf16 accumulators;
// A-fragments are reused across 4 column tiles (1.5 LDS loads per WMMA).
// MODE 0: silu + bf16 store (hidden).  MODE 1: f32 store + MSE accumulation.
// ---------------------------------------------------------------------------
template <int MODE>
__global__ __launch_bounds__(256)
void gemm_wmma(const unsigned short* __restrict__ A,
               const unsigned short* __restrict__ Bw,
               int K, int Jtot,
               unsigned short* __restrict__ outBf,
               float* __restrict__ outF,
               const float* __restrict__ xref,
               float* __restrict__ loss) {
    __shared__ __align__(16) unsigned short Asl[2][128 * 32];   // 2 x 8 KB
    __shared__ __align__(16) unsigned short Bsl[2][128 * 32];   // 2 x 8 KB
    __shared__ float red[256];

    const int tid  = threadIdx.x;
    const int i0   = blockIdx.y * 128;
    const int j0   = blockIdx.x * 128;
    const int wave = tid >> 5;
    const int lane = tid & 31;
    const int wi   = (wave & 3) * 32;     // wave row offset in block tile
    const int wj   = (wave >> 2) * 64;    // wave col offset in block tile
    const int lr   = lane & 15;
    const int hi   = lane >> 4;

    v8f acc[2][4];
#pragma unroll
    for (int ti = 0; ti < 2; ++ti)
#pragma unroll
        for (int tj = 0; tj < 4; ++tj)
            acc[ti][tj] = (v8f){0,0,0,0,0,0,0,0};

    union Frag { uint4 q[2]; v16bf v; };

    // Stage one 128x32 A tile + 128x32 B tile (each 512 x 16B; 2+2 per thread).
    auto stage = [&](int buf, int kk) {
#pragma unroll
        for (int r = 0; r < 2; ++r) {
            int L = tid + r * 256;
            int row = L >> 2, ko = (L & 3) * 8;
            *(uint4*)&Asl[buf][row * 32 + ko] =
                *(const uint4*)&A[(size_t)(i0 + row) * K + kk + ko];
            *(uint4*)&Bsl[buf][row * 32 + ko] =
                *(const uint4*)&Bw[(size_t)(j0 + row) * K + kk + ko];
        }
    };

    stage(0, 0);
    __syncthreads();

    int buf = 0;
    for (int kk = 32; kk <= K; kk += 32) {
        if (kk < K) stage(buf ^ 1, kk);    // prefetch into other buffer

        // A frag (ISA 16-bit 16x32 layout): lanes 0-15 -> K{0..7,16..23},
        // lanes 16-31 -> K{8..15,24..31}; row = lane%16.
        Frag a[2], b[4];
#pragma unroll
        for (int ti = 0; ti < 2; ++ti) {
            const unsigned short* ar = &Asl[buf][(wi + ti * 16 + lr) * 32];
            a[ti].q[0] = *(const uint4*)&ar[hi * 8];
            a[ti].q[1] = *(const uint4*)&ar[16 + hi * 8];
        }
        // B frag (32x16): lanes 0-15 -> K{0..7,16..23}, lanes 16-31 -> K{8..15,24..31},
        // col = lane%16; same K order as A in our J-major LDS tile.
#pragma unroll
        for (int tj = 0; tj < 4; ++tj) {
            const unsigned short* br = &Bsl[buf][(wj + tj * 16 + lr) * 32];
            b[tj].q[0] = *(const uint4*)&br[hi * 8];
            b[tj].q[1] = *(const uint4*)&br[16 + hi * 8];
        }

#pragma unroll
        for (int ti = 0; ti < 2; ++ti)
#pragma unroll
            for (int tj = 0; tj < 4; ++tj)
                acc[ti][tj] = __builtin_amdgcn_wmma_f32_16x16x32_bf16(
                    false, a[ti].v, false, b[tj].v, (short)0,
                    acc[ti][tj], false, false);

        __syncthreads();
        buf ^= 1;
    }

    // Epilogue. C layout: VGPR v, lanes 0-15 -> M=v, lanes 16-31 -> M=v+8.
    float lsum = 0.0f;
#pragma unroll
    for (int ti = 0; ti < 2; ++ti) {
#pragma unroll
        for (int tj = 0; tj < 4; ++tj) {
            int colg  = j0 + wj + tj * 16 + lr;
            int rbase = i0 + wi + ti * 16 + hi * 8;
#pragma unroll
            for (int v = 0; v < 8; ++v) {
                size_t idx = (size_t)(rbase + v) * Jtot + colg;
                float val = acc[ti][tj][v];
                if (MODE == 0) {
                    float sg = val / (1.0f + __expf(-val));   // silu
                    outBf[idx] = f2bf(sg);
                } else {
                    outF[idx] = val;
                    float df = val - xref[idx];
                    lsum += df * df;
                }
            }
        }
    }

    if (MODE == 1) {
        red[tid] = lsum;
        __syncthreads();
        for (int off = 128; off > 0; off >>= 1) {
            if (tid < off) red[tid] += red[tid + off];
            __syncthreads();
        }
        if (tid == 0)
            atomicAdd(loss, red[0] * (1.0f / ((float)NTOK * (float)D_)));
    }
}

// ---------------------------------------------------------------------------
extern "C" void kernel_launch(void* const* d_in, const int* in_sizes, int n_in,
                              void* d_out, int out_size, void* d_ws, size_t ws_size,
                              hipStream_t stream) {
    const float* x  = (const float*)d_in[0];   // [B,S,D]
    const float* w1 = (const float*)d_in[1];   // [M,D]
    const float* w2 = (const float*)d_in[2];   // [D,M]

    float* out    = (float*)d_out;
    float* memOut = out;                                   // [NTOK, D]
    float* loss   = out + (size_t)NTOK * D_;               // scalar
    float* state  = loss + 1;                              // [NTOK, D]

    char* ws = (char*)d_ws;
    size_t off = 0;
    unsigned short* stateBf  = (unsigned short*)(ws + off); off += (size_t)NTOK * D_ * 2;  // 32 MB
    unsigned short* hiddenBf = (unsigned short*)(ws + off); off += (size_t)NTOK * M_ * 2;  // 128 MB
    unsigned short* w1Bf     = (unsigned short*)(ws + off); off += (size_t)M_ * D_ * 2;    // 8 MB
    unsigned short* w2Bf     = (unsigned short*)(ws + off); off += (size_t)D_ * M_ * 2;    // 8 MB
    float*          carry    = (float*)(ws + off);                                         // 0.5 MB

    // EMA scan (3-pass chunked parallel linear scan)
    ema_pass1<<<(B_ * NCHUNK * D_) / 256, 256, 0, stream>>>(x, state, carry);
    ema_pass2<<<(B_ * D_) / 256, 256, 0, stream>>>(carry);
    ema_fixup<<<(int)(((size_t)NTOK * D_) / 256), 256, 0, stream>>>(state, carry, stateBf);

    // Weight down-conversion to bf16
    f32_to_bf16_kernel<<<(M_ * D_) / 256, 256, 0, stream>>>(w1, w1Bf, M_ * D_);
    f32_to_bf16_kernel<<<(D_ * M_) / 256, 256, 0, stream>>>(w2, w2Bf, D_ * M_);

    zero_loss<<<1, 1, 0, stream>>>(loss);

    // GEMM1: hidden = silu(state @ w1^T)   [16384 x 4096], K = 1024
    gemm_wmma<0><<<dim3(M_ / 128, NTOK / 128), 256, 0, stream>>>(
        stateBf, w1Bf, D_, M_, hiddenBf, nullptr, nullptr, nullptr);

    // GEMM2: mem_out = hidden @ w2^T       [16384 x 1024], K = 4096 (+ MSE)
    gemm_wmma<1><<<dim3(D_ / 128, NTOK / 128), 256, 0, stream>>>(
        hiddenBf, w2Bf, M_, D_, nullptr, memOut, x, loss);
}
